// Attention_5609227288590
// MI455X (gfx1250) — matrix operand order repaired and checked by reading, not compiled, not run on hardware
//
#include <hip/hip_runtime.h>

// ---------------- problem constants ----------------
#define DIMV   768
#define HEADS  12
#define DHEAD  64
#define INNER  768
#define NSEQ   1024
#define BATCH  16
#define QKV3   2304
#define MTOT   (BATCH * NSEQ)   // 16384 rows
#define QSCALE 0.125f           // 1/sqrt(64)

typedef __attribute__((ext_vector_type(16))) __bf16 v16bf;
typedef __attribute__((ext_vector_type(8)))  __bf16 v8bf;
typedef __attribute__((ext_vector_type(8)))  float  v8f;
typedef __attribute__((ext_vector_type(4)))  float  v4f;
typedef __attribute__((ext_vector_type(4)))  int    v4i;

#define BF16MMA(a, b, c) \
  __builtin_amdgcn_wmma_f32_16x16x32_bf16(false, (a), false, (b), (short)0, (c), false, false)

// Optional CDNA5 async global->LDS path
#if defined(__has_builtin)
#if __has_builtin(__builtin_amdgcn_global_load_async_to_lds_b128)
#define HAVE_ASYNC_LDS 1
#endif
#endif
#ifndef HAVE_ASYNC_LDS
#define HAVE_ASYNC_LDS 0
#endif

#if HAVE_ASYNC_LDS
#if __has_builtin(__builtin_amdgcn_s_wait_asynccnt)
#define WAIT_ASYNC(n) __builtin_amdgcn_s_wait_asynccnt(n)
#else
#define WAIT_ASYNC(n) asm volatile("s_wait_asynccnt %0" ::"i"(n) : "memory")
#endif
#endif

// ---------------- vectorized tile loaders (wave32, 16x16x32 bf16) ----------------
__device__ __forceinline__ v16bf ld16(const __bf16* p) {   // 32B contiguous per lane
  v8bf lo = *(const v8bf*)p;
  v8bf hi = *(const v8bf*)(p + 8);
  return __builtin_shufflevector(lo, hi, 0, 1, 2, 3, 4, 5, 6, 7, 8, 9, 10, 11, 12, 13, 14, 15);
}

// A-matrix 16x32 (MxK) from row-major bf16; lanes 0-15: K 0-7 & 16-23,
// lanes 16-31: K 8-15 & 24-31 (ISA 7.12.2). Two aligned 16B loads.
__device__ __forceinline__ v16bf load_a_bf16(const __bf16* A, int lda, int row0, int k0) {
  const int lane = threadIdx.x & 31;
  const __bf16* p = A + (size_t)(row0 + (lane & 15)) * lda + k0 + ((lane & 16) ? 8 : 0);
  v8bf lo = *(const v8bf*)p;
  v8bf hi = *(const v8bf*)(p + 16);
  return __builtin_shufflevector(lo, hi, 0, 1, 2, 3, 4, 5, 6, 7, 8, 9, 10, 11, 12, 13, 14, 15);
}

// A-matrix from fp32 source: four aligned float4 loads + cvt.
__device__ __forceinline__ v16bf load_a_f32(const float* A, int lda, int row0, int k0) {
  const int lane = threadIdx.x & 31;
  const float* p = A + (size_t)(row0 + (lane & 15)) * lda + k0 + ((lane & 16) ? 8 : 0);
  v4f x0 = *(const v4f*)p, x1 = *(const v4f*)(p + 4);
  v4f y0 = *(const v4f*)(p + 16), y1 = *(const v4f*)(p + 20);
  v16bf a;
#pragma unroll
  for (int i = 0; i < 4; ++i) {
    a[i] = (__bf16)x0[i];  a[4 + i]  = (__bf16)x1[i];
    a[8 + i] = (__bf16)y0[i]; a[12 + i] = (__bf16)y1[i];
  }
  return a;
}

// B-matrix 32x16 (KxN) where the contraction dim is CONTIGUOUS in memory:
// lanes 0-15: K 0-7 & 16-23, lanes 16-31: K 8-15 & 24-31 (two 16B reads per lane).
__device__ __forceinline__ v16bf load_bk(const __bf16* src, int ld, int c0, int k0) {
  const int lane = threadIdx.x & 31;
  const __bf16* p = src + (size_t)(c0 + (lane & 15)) * ld + k0 + ((lane & 16) ? 8 : 0);
  v8bf lo = *(const v8bf*)p;
  v8bf hi = *(const v8bf*)(p + 16);
  return __builtin_shufflevector(lo, hi, 0, 1, 2, 3, 4, 5, 6, 7, 8, 9, 10, 11, 12, 13, 14, 15);
}

// ---------------- kernel 0: weights fp32 -> bf16, TRANSPOSED ----------------
__global__ void cvt_kernel(const float* __restrict__ wqkv, const float* __restrict__ wout,
                           __bf16* __restrict__ wqT, __bf16* __restrict__ woT) {
  int i = blockIdx.x * blockDim.x + threadIdx.x;
  if (i < DIMV * QKV3) {
    int k = i / QKV3, c = i % QKV3;
    wqT[(size_t)c * DIMV + k] = (__bf16)wqkv[i];
  }
  if (i < INNER * DIMV) {
    int k = i / DIMV, c = i % DIMV;
    woT[(size_t)c * INNER + k] = (__bf16)wout[i];
  }
}

// ---------------- kernel 1: QKV projection ----------------
// Each wave: 16x64 output tile (4 wmma per K-chunk sharing one A load).
// q,k -> [B,H,N,D]; v -> transposed [B,H,D,N] (packed 16B stores).
__global__ __launch_bounds__(256) void qkv_kernel(const float* __restrict__ x,
                                                  const __bf16* __restrict__ wqT,
                                                  __bf16* __restrict__ qg,
                                                  __bf16* __restrict__ kg,
                                                  __bf16* __restrict__ vt) {
  const int t = blockIdx.x * 8 + (threadIdx.x >> 5);
  const int tm = t / (QKV3 / 64), tn = t % (QKV3 / 64);
  const int row0 = tm * 16, col0 = tn * 64;
  v8f acc[4] = {};
  for (int k0 = 0; k0 < DIMV; k0 += 32) {
    v16bf a = load_a_f32(x, DIMV, row0, k0);
#pragma unroll
    for (int u = 0; u < 4; ++u) {
      v16bf b = load_bk(wqT, DIMV, col0 + u * 16, k0);
      acc[u] = BF16MMA(a, b, acc[u]);
    }
  }
  const int lane = threadIdx.x & 31;
  const int which = col0 / INNER;        // 0=q 1=k 2=v (64-aligned: never straddles)
  const int h = (col0 % INNER) >> 6;     // head (64-aligned: single head)
  const int rb = (lane & 16) ? 8 : 0;
  const int bidx = row0 >> 10, n0 = (row0 & (NSEQ - 1)) + rb;
  if (which == 2) {                      // V transposed: rows are contiguous in n
#pragma unroll
    for (int u = 0; u < 4; ++u) {
      int d = u * 16 + (lane & 15);
      v8bf pk;
#pragma unroll
      for (int r = 0; r < 8; ++r) pk[r] = (__bf16)acc[u][r];
      *(v8bf*)&vt[(((size_t)bidx * HEADS + h) * DHEAD + d) * NSEQ + n0] = pk;
    }
  } else {
    __bf16* dst = (which == 0) ? qg : kg;
    const float sc = (which == 0) ? QSCALE : 1.0f;
#pragma unroll
    for (int u = 0; u < 4; ++u) {
      int d = u * 16 + (lane & 15);
#pragma unroll
      for (int r = 0; r < 8; ++r)
        dst[(((size_t)bidx * HEADS + h) * NSEQ + n0 + r) * DHEAD + d] = (__bf16)(acc[u][r] * sc);
    }
  }
}

#if HAVE_ASYNC_LDS
// Stage a 32(col) x 32(k) bf16 V panel into LDS: 4 async b128 per lane.
// Builtin signature (probed): (v4i AS(1)*, v4i AS(3)*, imm offset, imm cpol).
typedef __attribute__((address_space(1))) v4i* gv4p;
typedef __attribute__((address_space(3))) v4i* lv4p;

__device__ __forceinline__ void stage_v(const __bf16* vth, __bf16* lbuf, int dbase, int kk) {
  const int lane = threadIdx.x & 31;
  const __bf16* g = vth + (size_t)(dbase + lane) * NSEQ + kk;
  __bf16* l = lbuf + lane * 32;
  gv4p gg = (gv4p)g;
  lv4p ll = (lv4p)l;
#pragma unroll
  for (int q = 0; q < 4; ++q)
    __builtin_amdgcn_global_load_async_to_lds_b128(gg + q, ll + q, 0, 0);
}
#endif

// ---------------- kernel 2: fused attention ----------------
// One block (2 waves) = one (b,h) + 16 q-rows; 16x1024 score strip in LDS.
// Suppress (THRES=1): attn' = suppress * exp(2(s-m)) / (Z^2 * (1+1e-6)).
__global__ __launch_bounds__(64) void attn_kernel(const __bf16* __restrict__ qg,
                                                  const __bf16* __restrict__ kg,
                                                  const __bf16* __restrict__ vt,
                                                  __bf16* __restrict__ ao,
                                                  const float* __restrict__ suppress_p) {
  __shared__ float S[16 * NSEQ];   // 64 KB score/prob strip (CDNA5: 320KB LDS/WGP)
  __shared__ float redm[16][4];
  __shared__ float redz[16][4];
#if HAVE_ASYNC_LDS
  __shared__ __bf16 vstage[2][2][32][32];  // [buf][wave][col][k] = 8 KB
#endif
  const int bh   = blockIdx.x >> 6;
  const int row0 = (blockIdx.x & 63) * 16;
  const int wave = threadIdx.x >> 5;
  const int lane = threadIdx.x & 31;
  const __bf16* qp  = qg + (size_t)bh * NSEQ * DHEAD;
  const __bf16* kp  = kg + (size_t)bh * NSEQ * DHEAD;
  const __bf16* vth = vt + (size_t)bh * DHEAD * NSEQ;

  v16bf aq0 = load_a_bf16(qp, DHEAD, row0, 0);
  v16bf aq1 = load_a_bf16(qp, DHEAD, row0, 32);

  // phase 1: S = (q*scale) @ k^T  (32 columns per iteration per wave)
  for (int j = wave * 32; j < NSEQ; j += 64) {
    v16bf b00 = load_bk(kp, DHEAD, j, 0);
    v16bf b01 = load_bk(kp, DHEAD, j, 32);
    v16bf b10 = load_bk(kp, DHEAD, j + 16, 0);
    v16bf b11 = load_bk(kp, DHEAD, j + 16, 32);
    v8f s0 = {}, s1 = {};
    s0 = BF16MMA(aq0, b00, s0); s0 = BF16MMA(aq1, b01, s0);
    s1 = BF16MMA(aq0, b10, s1); s1 = BF16MMA(aq1, b11, s1);
    const int c = j + (lane & 15);
    const int rb = (lane & 16) ? 8 : 0;
#pragma unroll
    for (int r = 0; r < 8; ++r) {
      S[(rb + r) * NSEQ + c]      = s0[r];
      S[(rb + r) * NSEQ + c + 16] = s1[r];
    }
  }
  __syncthreads();

  // phase 2: row max + Z across 4 segments/row, rewrite strip as exp(2(s-m))
  {
    const int r2 = threadIdx.x >> 2, sg = threadIdx.x & 3;
    float* Sr = S + r2 * NSEQ + sg * 256;
    float m = -3.4e38f;
    for (int j = 0; j < 256; ++j) m = fmaxf(m, Sr[j]);
    redm[r2][sg] = m;
    __syncthreads();
    m = fmaxf(fmaxf(redm[r2][0], redm[r2][1]), fmaxf(redm[r2][2], redm[r2][3]));
    float z = 0.f;
    for (int j = 0; j < 256; ++j) {
      float e = __expf(Sr[j] - m);
      z += e;
      Sr[j] = e * e;                       // exp(2(s-m))
    }
    redz[r2][sg] = z;
  }
  __syncthreads();

  // phase 3: O = P @ V (32 output columns per wave), V panels via async->LDS
  v8f o0 = {}, o1 = {};
  const int dbase = wave * 32;
#if HAVE_ASYNC_LDS
  stage_v(vth, &vstage[0][wave][0][0], dbase, 0);
#endif
  for (int kk = 0; kk < NSEQ; kk += 32) {
    const int cur = (kk >> 5) & 1;
#if HAVE_ASYNC_LDS
    if (kk + 32 < NSEQ) {
      stage_v(vth, &vstage[cur ^ 1][wave][0][0], dbase, kk + 32);
      WAIT_ASYNC(4);                       // first panel's 4 copies done (in-order)
    } else {
      WAIT_ASYNC(0);
    }
#endif
    const int rr = lane & 15;
    const int kb = (lane & 16) ? 8 : 0;
    const float* Pr = S + rr * NSEQ + kk + kb;
    v16bf a;
#pragma unroll
    for (int i = 0; i < 8; ++i) { a[i] = (__bf16)Pr[i]; a[i + 8] = (__bf16)Pr[16 + i]; }
#if HAVE_ASYNC_LDS
    const __bf16* vb = &vstage[cur][wave][0][0];
    const int ko = (lane & 16) ? 8 : 0;
    const __bf16* pb0 = vb + (size_t)((lane & 15)) * 32 + ko;
    const __bf16* pb1 = vb + (size_t)((lane & 15) + 16) * 32 + ko;
    v16bf b0 = __builtin_shufflevector(*(const v8bf*)pb0, *(const v8bf*)(pb0 + 16), 0, 1, 2, 3, 4, 5, 6, 7, 8, 9, 10, 11, 12, 13, 14, 15);
    v16bf b1 = __builtin_shufflevector(*(const v8bf*)pb1, *(const v8bf*)(pb1 + 16), 0, 1, 2, 3, 4, 5, 6, 7, 8, 9, 10, 11, 12, 13, 14, 15);
#else
    v16bf b0 = load_bk(vth, NSEQ, dbase, kk);
    v16bf b1 = load_bk(vth, NSEQ, dbase + 16, kk);
#endif
    o0 = BF16MMA(a, b0, o0);
    o1 = BF16MMA(a, b1, o1);
  }

  const float sup = *suppress_p;
  const int bidx = bh / HEADS, h = bh % HEADS;
  const int c0 = dbase + (lane & 15);
  const int rb = (lane & 16) ? 8 : 0;
#pragma unroll
  for (int r = 0; r < 8; ++r) {
    const int rw = rb + r;
    float z = (redz[rw][0] + redz[rw][1]) + (redz[rw][2] + redz[rw][3]);
    float sc = sup / (z * z * 1.000001f);
    size_t base = ((size_t)bidx * NSEQ + row0 + rw) * INNER + (size_t)h * DHEAD;
    ao[base + c0]      = (__bf16)(o0[r] * sc);
    ao[base + c0 + 16] = (__bf16)(o1[r] * sc);
  }
}

// ---------------- kernel 3: output projection + bias ----------------
__global__ __launch_bounds__(256) void proj_kernel(const __bf16* __restrict__ ao,
                                                   const __bf16* __restrict__ woT,
                                                   const float* __restrict__ bias,
                                                   float* __restrict__ out) {
  const int t = blockIdx.x * 8 + (threadIdx.x >> 5);
  const int tm = t / (DIMV / 64), tn = t % (DIMV / 64);
  const int row0 = tm * 16, col0 = tn * 64;
  v8f acc[4] = {};
  for (int k0 = 0; k0 < INNER; k0 += 32) {
    v16bf a = load_a_bf16(ao, INNER, row0, k0);
#pragma unroll
    for (int u = 0; u < 4; ++u) {
      v16bf b = load_bk(woT, INNER, col0 + u * 16, k0);
      acc[u] = BF16MMA(a, b, acc[u]);
    }
  }
  const int lane = threadIdx.x & 31;
  const int rb = (lane & 16) ? 8 : 0;
#pragma unroll
  for (int u = 0; u < 4; ++u) {
    const int c = col0 + u * 16 + (lane & 15);
    const float bb = bias[c];
#pragma unroll
    for (int r = 0; r < 8; ++r)
      out[(size_t)(row0 + rb + r) * DIMV + c] = acc[u][r] + bb;
  }
}

// ---------------- launcher ----------------
extern "C" void kernel_launch(void* const* d_in, const int* in_sizes, int n_in,
                              void* d_out, int out_size, void* d_ws, size_t ws_size,
                              hipStream_t stream) {
  (void)in_sizes; (void)n_in; (void)out_size; (void)ws_size;
  const float* x        = (const float*)d_in[0];
  const float* w_qkv    = (const float*)d_in[1];
  const float* w_out    = (const float*)d_in[2];
  const float* b_out    = (const float*)d_in[3];
  const float* suppress = (const float*)d_in[4];

  char* p = (char*)d_ws;
  __bf16* wqT = (__bf16*)p; p += (size_t)DIMV * QKV3 * 2;   // 3.5 MB, [2304][768]
  __bf16* woT = (__bf16*)p; p += (size_t)INNER * DIMV * 2;  // 1.2 MB, [768][768]
  const size_t hsz = (size_t)BATCH * HEADS * NSEQ * DHEAD;
  __bf16* qg = (__bf16*)p; p += hsz * 2;                    // [B,H,N,D]
  __bf16* kg = (__bf16*)p; p += hsz * 2;                    // [B,H,N,D]
  __bf16* vt = (__bf16*)p; p += hsz * 2;                    // [B,H,D,N] (transposed)
  __bf16* ao = (__bf16*)p; p += (size_t)MTOT * INNER * 2;   // [M, INNER]

  const int ncvt = DIMV * QKV3;
  cvt_kernel<<<(ncvt + 255) / 256, 256, 0, stream>>>(w_qkv, w_out, wqT, woT);
  qkv_kernel<<<(MTOT / 16) * (QKV3 / 64) / 8, 256, 0, stream>>>(x, wqT, qg, kg, vt);
  attn_kernel<<<BATCH * HEADS * (NSEQ / 16), 64, 0, stream>>>(qg, kg, vt, ao, suppress);
  proj_kernel<<<(MTOT / 16) * (DIMV / 64) / 8, 256, 0, stream>>>(ao, woT, b_out, (float*)d_out);
}
